// TransformerDecoderLayer_53429393162533
// MI455X (gfx1250) — matrix operand order repaired and checked by reading, not compiled, not run on hardware
//
#include <hip/hip_runtime.h>
#include <hip/hip_bf16.h>

// ---------------------------------------------------------------------------
// TransformerDecoderLayer on MI455X (gfx1250): f16 WMMA compute, f32 accum.
// B=4, T=S=2048, D=1024, F=4096, H=16, Hd=64.
// GEMMs: double-buffered LDS, A-tile staged via global_load_async_to_lds_b128
// (ASYNCcnt), B-tile transposed through VGPRs, 16x16x32 f16 WMMA.
// ---------------------------------------------------------------------------

typedef __attribute__((ext_vector_type(16))) _Float16 v16h;
typedef __attribute__((ext_vector_type(8)))  float    v8f;

union Frag16 { v16h h; unsigned u[8]; _Float16 e[16]; };

static __device__ __forceinline__ v8f wmma_f16(v16h a, v16h b, v8f c) {
  // D = A(16x32 f16) * B(32x16 f16) + C(16x16 f32)
  return __builtin_amdgcn_wmma_f32_16x16x32_f16(
      /*neg_a=*/false, a, /*neg_b=*/false, b,
      /*c_mod=*/(short)0, c, /*reuse_a=*/false, /*reuse_b=*/false);
}

#define DCONST 1024
#define HCONST 16
#define HDCONST 64

// ---------------------------------------------------------------------------
// elementwise f32 -> f16 conversion
// ---------------------------------------------------------------------------
__global__ void cvt_f32_f16_kernel(const float* __restrict__ in,
                                   _Float16* __restrict__ out, int n) {
  int i = blockIdx.x * blockDim.x + threadIdx.x;
  int stride = gridDim.x * blockDim.x;
  for (; i < n; i += stride) out[i] = (_Float16)in[i];
}

// ---------------------------------------------------------------------------
// LayerNorm over D=1024, one block per row, f32 in -> f16 out
// ---------------------------------------------------------------------------
__global__ __launch_bounds__(256) void ln_kernel(const float* __restrict__ x,
                                                 const float* __restrict__ g,
                                                 const float* __restrict__ bta,
                                                 _Float16* __restrict__ out) {
  __shared__ float red[256];
  int row = blockIdx.x;
  const float* xr = x + (size_t)row * DCONST;
  float s = 0.f, s2 = 0.f;
  for (int i = threadIdx.x; i < DCONST; i += 256) {
    float v = xr[i];
    s += v; s2 += v * v;
  }
  red[threadIdx.x] = s;
  __syncthreads();
  for (int k = 128; k > 0; k >>= 1) {
    if (threadIdx.x < k) red[threadIdx.x] += red[threadIdx.x + k];
    __syncthreads();
  }
  float mean = red[0] * (1.f / DCONST);
  __syncthreads();
  red[threadIdx.x] = s2;
  __syncthreads();
  for (int k = 128; k > 0; k >>= 1) {
    if (threadIdx.x < k) red[threadIdx.x] += red[threadIdx.x + k];
    __syncthreads();
  }
  float var = red[0] * (1.f / DCONST) - mean * mean;
  float inv = rsqrtf(var + 1e-5f);
  for (int i = threadIdx.x; i < DCONST; i += 256) {
    float v = (xr[i] - mean) * inv * g[i] + bta[i];
    out[(size_t)row * DCONST + i] = (_Float16)v;
  }
}

// ---------------------------------------------------------------------------
// GEMM: out[M,N] = A[M,K](f16) @ Bw[K,N](f16) + bias[N] (+resid) (+relu)
// block tile 128x128, 8 waves, each wave 64x32 (4x2 WMMA tiles).
// Double-buffered LDS; A staged with async global->LDS (ASYNCcnt).
// ---------------------------------------------------------------------------
template <int OF16, int RELU, int RES>
__global__ __launch_bounds__(256) void gemm_kernel(
    const _Float16* __restrict__ A, const _Float16* __restrict__ Bw,
    const float* __restrict__ bias, const float* __restrict__ resid,
    void* __restrict__ outp, int M, int N, int K) {
  __shared__ __align__(16) _Float16 sA[2][128 * 32];  // [m][k] stride 32
  __shared__ __align__(16) _Float16 sB[2][128 * 34];  // transposed [n][k] str 34

  int t = threadIdx.x;
  int m0 = blockIdx.y * 128, n0 = blockIdx.x * 128;
  int lane = t & 31, w = t >> 5;
  int wm = w >> 2, wn = w & 3;  // wave grid 2x4
  int c16 = lane & 15, lh = lane >> 4;

  const v8f vz = {0.f, 0.f, 0.f, 0.f, 0.f, 0.f, 0.f, 0.f};
  v8f acc[4][2];
#pragma unroll
  for (int a = 0; a < 4; a++)
#pragma unroll
    for (int bq = 0; bq < 2; bq++) acc[a][bq] = vz;

  auto stage = [&](int kk, int buf) {
    // A tile 128x32: async copy global -> LDS (no VGPR round trip)
#pragma unroll
    for (int it = 0; it < 2; ++it) {
      int row = (t >> 2) + it * 64;
      int ch = (t & 3) * 8;
      unsigned lofs = (unsigned)(uintptr_t)&sA[buf][row * 32 + ch];
      unsigned long long ga =
          (unsigned long long)(uintptr_t)(A + (size_t)(m0 + row) * K + kk + ch);
      asm volatile("global_load_async_to_lds_b128 %0, %1, off"
                   :: "v"(lofs), "v"(ga)
                   : "memory");
    }
    // B tile 32x128: load coalesced, store transposed [n][k]
#pragma unroll
    for (int it = 0; it < 2; ++it) {
      int kr = (t >> 4) + it * 16;
      int ch = (t & 15) * 8;
      union { uint4 v; _Float16 hh[8]; } tb;
      tb.v = *(const uint4*)(Bw + (size_t)(kk + kr) * N + n0 + ch);
#pragma unroll
      for (int j = 0; j < 8; j++) sB[buf][(ch + j) * 34 + kr] = tb.hh[j];
    }
  };

  int nk = K >> 5;
  stage(0, 0);
  for (int i = 0; i < nk; ++i) {
    int cur = i & 1;
    asm volatile("s_wait_asynccnt 0x0" ::: "memory");
    __syncthreads();  // staged data (async A + ds B) visible to all waves
    if (i + 1 < nk) stage((i + 1) << 5, cur ^ 1);  // overlap next stage

    Frag16 af[4], bf[2];
#pragma unroll
    for (int mt = 0; mt < 4; ++mt) {
      int m = wm * 64 + mt * 16 + c16;
#pragma unroll
      for (int v = 0; v < 8; ++v) {
        int k0 = (v < 4 ? 2 * v + lh * 8 : 2 * v + 8 + lh * 8);
        af[mt].u[v] = *(const unsigned*)&sA[cur][m * 32 + k0];
      }
    }
#pragma unroll
    for (int nt = 0; nt < 2; ++nt) {
      int n = wn * 32 + nt * 16 + c16;
#pragma unroll
      for (int v = 0; v < 8; ++v) {
        int k0 = (v < 4 ? 2 * v + lh * 8 : 2 * v + 8 + lh * 8);
        bf[nt].u[v] = *(const unsigned*)&sB[cur][n * 34 + k0];
      }
    }
#pragma unroll
    for (int mt = 0; mt < 4; ++mt)
#pragma unroll
      for (int nt = 0; nt < 2; ++nt)
        acc[mt][nt] = wmma_f16(af[mt].h, bf[nt].h, acc[mt][nt]);
    __syncthreads();  // all waves done reading cur before it is restaged
  }

  // epilogue: bias (+resid) (+relu), f16 or f32 out
#pragma unroll
  for (int mt = 0; mt < 4; ++mt) {
#pragma unroll
    for (int nt = 0; nt < 2; ++nt) {
      int col = n0 + wn * 32 + nt * 16 + c16;
      float bv = bias[col];
#pragma unroll
      for (int r = 0; r < 8; ++r) {
        int row = m0 + wm * 64 + mt * 16 + r + lh * 8;
        float vout = acc[mt][nt][r] + bv;
        if (RES) vout += resid[(size_t)row * N + col];
        if (RELU) vout = fmaxf(vout, 0.f);
        if (OF16)
          ((_Float16*)outp)[(size_t)row * N + col] = (_Float16)vout;
        else
          ((float*)outp)[(size_t)row * N + col] = vout;
      }
    }
  }
}

// ---------------------------------------------------------------------------
// Flash attention: one wave per (b, head, 16-row q tile). f16 QKV, f32 accum.
// ---------------------------------------------------------------------------
__global__ __launch_bounds__(32) void attn_kernel(
    const _Float16* __restrict__ Q, const _Float16* __restrict__ Km,
    const _Float16* __restrict__ Vm, _Float16* __restrict__ O, int Bc, int T,
    int S, int causal) {
  __shared__ __align__(16) _Float16 ldsP[16 * 34];

  int tilesT = T / 16;
  int bid = blockIdx.x;
  int tile = bid % tilesT;
  int hq = (bid / tilesT) % HCONST;
  int b = bid / (tilesT * HCONST);
  int t0 = tile * 16;
  int lane = threadIdx.x, c16 = lane & 15, lh = lane >> 4;

  const _Float16* Qb = Q + ((size_t)b * T) * DCONST + hq * HDCONST;
  const _Float16* Kb = Km + ((size_t)b * S) * DCONST + hq * HDCONST;
  const _Float16* Vb = Vm + ((size_t)b * S) * DCONST + hq * HDCONST;

  // preload Q tile 16x64 as two A fragments (Hd = 2 * 32)
  Frag16 qa[2];
#pragma unroll
  for (int kc = 0; kc < 2; ++kc)
#pragma unroll
    for (int v = 0; v < 8; ++v) {
      int k0 = (v < 4 ? 2 * v + lh * 8 : 2 * v + 8 + lh * 8) + kc * 32;
      qa[kc].u[v] = *(const unsigned*)(Qb + (size_t)(t0 + c16) * DCONST + k0);
    }

  const v8f vz = {0.f, 0.f, 0.f, 0.f, 0.f, 0.f, 0.f, 0.f};
  v8f oacc[4];
#pragma unroll
  for (int j = 0; j < 4; j++) oacc[j] = vz;
  float mrun[8], lrun[8];
#pragma unroll
  for (int r = 0; r < 8; r++) { mrun[r] = -1e30f; lrun[r] = 0.f; }

  int smax = causal ? (t0 + 16) : S;
  for (int s0 = 0; s0 < smax; s0 += 32) {
    // ---- scores: two 16x16 f32 tiles (columns s0..s0+15, s0+16..s0+31) ----
    v8f st[2];
#pragma unroll
    for (int g = 0; g < 2; ++g) {
      Frag16 kf0, kf1;
      size_t srow = (size_t)(s0 + g * 16 + c16);
#pragma unroll
      for (int v = 0; v < 8; ++v) {
        int k0 = (v < 4 ? 2 * v + lh * 8 : 2 * v + 8 + lh * 8);
        kf0.u[v] = *(const unsigned*)(Kb + srow * DCONST + k0);
        kf1.u[v] = *(const unsigned*)(Kb + srow * DCONST + 32 + k0);
      }
      v8f z = vz;
      z = wmma_f16(qa[0].h, kf0.h, z);
      z = wmma_f16(qa[1].h, kf1.h, z);
      st[g] = z;
    }
    // ---- online softmax (row stats per lane-half: rows lh*8 + r) ----
    float p0[8], p1[8];
#pragma unroll
    for (int r = 0; r < 8; ++r) {
      float a = st[0][r] * 0.125f;  // 1/sqrt(64)
      float c = st[1][r] * 0.125f;
      if (causal) {
        int trow = t0 + r + lh * 8;
        if (s0 + c16 > trow) a = -1e30f;
        if (s0 + 16 + c16 > trow) c = -1e30f;
      }
      float mx = fmaxf(a, c);
#pragma unroll
      for (int msk = 1; msk < 16; msk <<= 1)
        mx = fmaxf(mx, __shfl_xor(mx, msk, 32));
      float newm = fmaxf(mrun[r], mx);
      float corr = __expf(mrun[r] - newm);
      float e0 = __expf(a - newm);
      float e1 = __expf(c - newm);
      float rs = e0 + e1;
#pragma unroll
      for (int msk = 1; msk < 16; msk <<= 1) rs += __shfl_xor(rs, msk, 32);
      lrun[r] = lrun[r] * corr + rs;
      mrun[r] = newm;
#pragma unroll
      for (int j = 0; j < 4; ++j) oacc[j][r] *= corr;
      p0[r] = e0; p1[r] = e1;
    }
    // ---- transpose P (C-layout -> A-layout) through LDS ----
#pragma unroll
    for (int r = 0; r < 8; ++r) {
      ldsP[(r + lh * 8) * 34 + c16] = (_Float16)p0[r];
      ldsP[(r + lh * 8) * 34 + 16 + c16] = (_Float16)p1[r];
    }
    __syncthreads();
    Frag16 pf;
#pragma unroll
    for (int v = 0; v < 8; ++v) {
      int k0 = 2 * v + lh * 16;
      pf.u[v] = *(const unsigned*)&ldsP[c16 * 34 + k0];
    }
    // ---- O += P(16x32) @ V(32x64) : 4 WMMAs ----
#pragma unroll
    for (int j = 0; j < 4; ++j) {
      Frag16 vf;
#pragma unroll
      for (int e = 0; e < 16; ++e) {
        int k = e + lh * 16;
        vf.e[e] = Vb[(size_t)(s0 + k) * DCONST + j * 16 + c16];
      }
      oacc[j] = wmma_f16(pf.h, vf.h, oacc[j]);
    }
    __syncthreads();
  }

  _Float16* Ob = O + ((size_t)b * T) * DCONST + hq * HDCONST;
#pragma unroll
  for (int j = 0; j < 4; ++j)
#pragma unroll
    for (int r = 0; r < 8; ++r) {
      float val = oacc[j][r] / lrun[r];
      Ob[(size_t)(t0 + r + lh * 8) * DCONST + j * 16 + c16] = (_Float16)val;
    }
}

// ---------------------------------------------------------------------------
// host orchestration
// ---------------------------------------------------------------------------
template <int OF16, int RELU, int RES>
static void run_gemm(const _Float16* A, const _Float16* Bw, const float* bias,
                     const float* resid, void* out, int M, int N, int K,
                     hipStream_t s) {
  dim3 grid(N / 128, M / 128);
  gemm_kernel<OF16, RELU, RES><<<grid, 256, 0, s>>>(A, Bw, bias, resid, out, M, N, K);
}

static void run_cvt(const float* in, _Float16* out, int n, hipStream_t s) {
  int blocks = (n + 255) / 256;
  if (blocks > 4096) blocks = 4096;
  cvt_f32_f16_kernel<<<blocks, 256, 0, s>>>(in, out, n);
}

extern "C" void kernel_launch(void* const* d_in, const int* in_sizes, int n_in,
                              void* d_out, int out_size, void* d_ws,
                              size_t ws_size, hipStream_t stream) {
  (void)in_sizes; (void)n_in; (void)out_size; (void)ws_size;
  const int Bc = 4, T = 2048, S = 2048, D = 1024, F = 4096;
  const int M = Bc * T;   // 8192 query rows
  const int Ms = Bc * S;  // 8192 kv rows

  const float* x      = (const float*)d_in[0];
  const float* enc    = (const float*)d_in[1];
  const float* sa_qw  = (const float*)d_in[2];
  const float* sa_qb  = (const float*)d_in[3];
  const float* sa_kw  = (const float*)d_in[4];
  const float* sa_kb  = (const float*)d_in[5];
  const float* sa_vw  = (const float*)d_in[6];
  const float* sa_vb  = (const float*)d_in[7];
  const float* sa_ow  = (const float*)d_in[8];
  const float* sa_ob  = (const float*)d_in[9];
  const float* ca_qw  = (const float*)d_in[10];
  const float* ca_qb  = (const float*)d_in[11];
  const float* ca_kw  = (const float*)d_in[12];
  const float* ca_kb  = (const float*)d_in[13];
  const float* ca_vw  = (const float*)d_in[14];
  const float* ca_vb  = (const float*)d_in[15];
  const float* ca_ow  = (const float*)d_in[16];
  const float* ca_ob  = (const float*)d_in[17];
  const float* ff_w1  = (const float*)d_in[18];
  const float* ff_b1  = (const float*)d_in[19];
  const float* ff_w2  = (const float*)d_in[20];
  const float* ff_b2  = (const float*)d_in[21];
  const float* ln1_g  = (const float*)d_in[22];
  const float* ln1_b  = (const float*)d_in[23];
  const float* ln2_g  = (const float*)d_in[24];
  const float* ln2_b  = (const float*)d_in[25];
  const float* ln3_g  = (const float*)d_in[26];
  const float* ln3_b  = (const float*)d_in[27];

  char* ws = (char*)d_ws;
  size_t off = 0;
  auto alloc = [&](size_t bytes) -> void* {
    void* p = ws + off;
    off += (bytes + 255) & ~(size_t)255;
    return p;
  };
  const size_t DD = (size_t)D * D, DF = (size_t)D * F;
  _Float16* w_saq = (_Float16*)alloc(DD * 2);
  _Float16* w_sak = (_Float16*)alloc(DD * 2);
  _Float16* w_sav = (_Float16*)alloc(DD * 2);
  _Float16* w_sao = (_Float16*)alloc(DD * 2);
  _Float16* w_caq = (_Float16*)alloc(DD * 2);
  _Float16* w_cak = (_Float16*)alloc(DD * 2);
  _Float16* w_cav = (_Float16*)alloc(DD * 2);
  _Float16* w_cao = (_Float16*)alloc(DD * 2);
  _Float16* w_ff1 = (_Float16*)alloc(DF * 2);
  _Float16* w_ff2 = (_Float16*)alloc(DF * 2);
  _Float16* enc16 = (_Float16*)alloc((size_t)Ms * D * 2);
  _Float16* h16   = (_Float16*)alloc((size_t)M * D * 2);
  _Float16* q16   = (_Float16*)alloc((size_t)M * D * 2);
  _Float16* k16   = (_Float16*)alloc((size_t)Ms * D * 2);
  _Float16* v16   = (_Float16*)alloc((size_t)Ms * D * 2);
  _Float16* ao16  = (_Float16*)alloc((size_t)M * D * 2);
  float*    x1    = (float*)alloc((size_t)M * D * 4);
  float*    x2    = (float*)alloc((size_t)M * D * 4);
  _Float16* ffn1  = (_Float16*)alloc((size_t)M * F * 2);

  // one-time f32 -> f16 conversions
  run_cvt(sa_qw, w_saq, (int)DD, stream);
  run_cvt(sa_kw, w_sak, (int)DD, stream);
  run_cvt(sa_vw, w_sav, (int)DD, stream);
  run_cvt(sa_ow, w_sao, (int)DD, stream);
  run_cvt(ca_qw, w_caq, (int)DD, stream);
  run_cvt(ca_kw, w_cak, (int)DD, stream);
  run_cvt(ca_vw, w_cav, (int)DD, stream);
  run_cvt(ca_ow, w_cao, (int)DD, stream);
  run_cvt(ff_w1, w_ff1, (int)DF, stream);
  run_cvt(ff_w2, w_ff2, (int)DF, stream);
  run_cvt(enc, enc16, Ms * D, stream);

  // ---- self attention block ----
  ln_kernel<<<M, 256, 0, stream>>>(x, ln1_g, ln1_b, h16);
  run_gemm<1, 0, 0>(h16, w_saq, sa_qb, nullptr, q16, M, D, D, stream);
  run_gemm<1, 0, 0>(h16, w_sak, sa_kb, nullptr, k16, M, D, D, stream);
  run_gemm<1, 0, 0>(h16, w_sav, sa_vb, nullptr, v16, M, D, D, stream);
  attn_kernel<<<Bc * HCONST * (T / 16), 32, 0, stream>>>(q16, k16, v16, ao16,
                                                         Bc, T, T, 1);
  run_gemm<0, 0, 1>(ao16, w_sao, sa_ob, x, x1, M, D, D, stream);

  // ---- cross attention block ----
  ln_kernel<<<M, 256, 0, stream>>>(x1, ln2_g, ln2_b, h16);
  run_gemm<1, 0, 0>(h16, w_caq, ca_qb, nullptr, q16, M, D, D, stream);
  run_gemm<1, 0, 0>(enc16, w_cak, ca_kb, nullptr, k16, Ms, D, D, stream);
  run_gemm<1, 0, 0>(enc16, w_cav, ca_vb, nullptr, v16, Ms, D, D, stream);
  attn_kernel<<<Bc * HCONST * (T / 16), 32, 0, stream>>>(q16, k16, v16, ao16,
                                                         Bc, T, S, 0);
  run_gemm<0, 0, 1>(ao16, w_cao, ca_ob, x1, x2, M, D, D, stream);

  // ---- FFN block ----
  ln_kernel<<<M, 256, 0, stream>>>(x2, ln3_g, ln3_b, h16);
  run_gemm<1, 1, 0>(h16, w_ff1, ff_b1, nullptr, ffn1, M, F, D, stream);
  run_gemm<0, 0, 1>(ffn1, w_ff2, ff_b2, x2, (float*)d_out, M, D, F, stream);
}
